// MLAAttention_84782654423898
// MI455X (gfx1250) — matrix operand order repaired and checked by reading, not compiled, not run on hardware
//
#include <hip/hip_runtime.h>
#include <hip/hip_bf16.h>
#include <math.h>

// ---------------------------------------------------------------------------
// MLA attention for gfx1250 (CDNA5, wave32, WMMA 16x16x32 f16 -> f32 accum)
// GEMMs: double-buffered LDS pipeline with async global->LDS copies.
// Attention: flash-attention; K tiles staged by the Tensor Data Mover.
// ---------------------------------------------------------------------------

typedef __attribute__((ext_vector_type(16))) _Float16 v16h;
typedef __attribute__((ext_vector_type(8)))  float    v8f;
typedef unsigned int u32x4 __attribute__((ext_vector_type(4)));
typedef int          i32x8 __attribute__((ext_vector_type(8)));
typedef int          i32x4 __attribute__((ext_vector_type(4)));
typedef int          v4i_g __attribute__((vector_size(16)));   // 4 x i32 (GCC-style)

#define HIDDEN   2048
#define SEQ      2048
#define NBATCH   2
#define NHEADS   16
#define HEAD_DIM 128
#define KV_DIM   128
#define SLOPE    0.0625f

#ifndef __has_builtin
#define __has_builtin(x) 0
#endif

#if __has_builtin(__builtin_amdgcn_global_load_async_to_lds_b128)
#define HAVE_ASYNC_LDS 1
#else
#define HAVE_ASYNC_LDS 0
#endif

#if __has_builtin(__builtin_amdgcn_tensor_load_to_lds) && \
    __has_builtin(__builtin_amdgcn_s_wait_tensorcnt)
#define HAVE_TDM 1
#else
#define HAVE_TDM 0
#endif

// Builtin parameter types per hipcc diagnostics:
//   arg0: v4i* in AS1 ("__device__"), non-const; arg1: v4i* in AS3.
#define GLOBAL_AS_V4(p) ((__attribute__((address_space(1))) v4i_g*)(p))
#define LDS_AS_V4(p)    ((__attribute__((address_space(3))) v4i_g*)(p))

__device__ inline void wait_async0() {
#if HAVE_ASYNC_LDS
#if __has_builtin(__builtin_amdgcn_s_wait_asynccnt)
  __builtin_amdgcn_s_wait_asynccnt(0);
#else
  asm volatile("s_wait_asynccnt 0x0" ::: "memory");
#endif
#endif
}

// Stage one 16-byte chunk global -> LDS.
__device__ inline void stage16(const _Float16* gsrc, _Float16* ldst) {
#if HAVE_ASYNC_LDS
  __builtin_amdgcn_global_load_async_to_lds_b128(GLOBAL_AS_V4(gsrc), LDS_AS_V4(ldst), 0, 0);
#else
  *(uint4*)ldst = *(const uint4*)gsrc;
#endif
}

union FragU { v16h v; uint4 u[2]; };

// A-matrix fragment (16x32 f16, MxK): lane m = lane&15 holds row m.
// halves 0..15 -> K = kOff + base + 0..15     (base = 0 for lanes<16, 16 otherwise)
__device__ inline v16h load_fragA(const _Float16* p, int rowStride, int rowBase,
                                  int kOff, int lane) {
  const int m    = lane & 15;
  const int base = (lane >> 4) << 4;  // 0 or 16
  const _Float16* row = p + (size_t)(rowBase + m) * rowStride + kOff;
  FragU f;
  f.u[0] = *(const uint4*)(row + base);
  f.u[1] = *(const uint4*)(row + base + 8);
  return f.v;
}

// B-matrix fragment (32x16 f16, KxN), stored as B^T rows (N x K row-major):
// lane n = lane&15 holds column n; halves 0..15 -> K = kOff + (lane>=16?16:0) + 0..15
__device__ inline v16h load_fragB(const _Float16* p, int rowStride, int rowBase,
                                  int kOff, int lane) {
  const int n    = lane & 15;
  const int base = (lane >> 4) << 4;  // 0 or 16
  const _Float16* row = p + (size_t)(rowBase + n) * rowStride + kOff + base;
  FragU f;
  f.u[0] = *(const uint4*)(row);
  f.u[1] = *(const uint4*)(row + 8);
  return f.v;
}

// ---------------------------------------------------------------------------
// Generic C[M,N] = A[M,K] @ B[N,K]^T, f16 inputs, f32 accum, OutT output.
// Block: 256 threads = 8 waves; tile 128x128, K-step 32, double-buffered LDS.
// Wave (w/2, w&1) computes a 32x64 sub-tile = 2x4 grid of 16x16 WMMAs.
// M, N multiples of 128; K multiple of 32 (true for all calls here).
// ---------------------------------------------------------------------------
template <typename OutT>
__global__ __launch_bounds__(256) void gemm_f16_wmma(
    const _Float16* __restrict__ A, const _Float16* __restrict__ B,
    OutT* __restrict__ C, int M, int N, int K) {
  __shared__ alignas(16) _Float16 As[2][128 * 32];
  __shared__ alignas(16) _Float16 Bs[2][128 * 32];

  const int tid  = threadIdx.x;
  const int lane = tid & 31;
  const int wave = tid >> 5;
  const int mBase = blockIdx.y * 128;
  const int nBase = blockIdx.x * 128;
  const int wM = (wave >> 1) * 32;  // 0,32,64,96
  const int wN = (wave & 1) * 64;   // 0,64

  // This thread's two 16B staging chunks per 128x32 tile.
  const int row0  = (tid + 0)   >> 2;        // 0..63
  const int part0 = ((tid + 0)  & 3) * 8;
  const int row1  = (tid + 256) >> 2;        // 64..127
  const int part1 = ((tid + 256) & 3) * 8;

  auto stage = [&](int buf, int kb) {
    stage16(A + (size_t)(mBase + row0) * K + kb + part0, &As[buf][row0 * 32 + part0]);
    stage16(B + (size_t)(nBase + row0) * K + kb + part0, &Bs[buf][row0 * 32 + part0]);
    stage16(A + (size_t)(mBase + row1) * K + kb + part1, &As[buf][row1 * 32 + part1]);
    stage16(B + (size_t)(nBase + row1) * K + kb + part1, &Bs[buf][row1 * 32 + part1]);
  };

  v8f zero = {};
  v8f acc[2][4];
  for (int i = 0; i < 2; ++i)
    for (int j = 0; j < 4; ++j) acc[i][j] = zero;

  stage(0, 0);
  wait_async0();
  __syncthreads();

  int cur = 0;
  for (int kb = 0; kb < K; kb += 32) {
    if (kb + 32 < K) stage(cur ^ 1, kb + 32);  // prefetch next tile into other buffer

    v16h aF[2], bF[4];
#pragma unroll
    for (int i = 0; i < 2; ++i) aF[i] = load_fragA(As[cur], 32, wM + i * 16, 0, lane);
#pragma unroll
    for (int j = 0; j < 4; ++j) bF[j] = load_fragB(Bs[cur], 32, wN + j * 16, 0, lane);
#pragma unroll
    for (int i = 0; i < 2; ++i)
#pragma unroll
      for (int j = 0; j < 4; ++j)
        acc[i][j] = __builtin_amdgcn_wmma_f32_16x16x32_f16(
            false, aF[i], false, bF[j], (short)0, acc[i][j], false, false);

    wait_async0();
    __syncthreads();
    cur ^= 1;
  }

  // Epilogue: C/D layout row = r + 8*(lane>=16), col = lane&15.
  const int rsel = (lane >> 4) << 3;
  const int col0 = lane & 15;
  for (int i = 0; i < 2; ++i)
    for (int j = 0; j < 4; ++j) {
      int col  = nBase + wN + j * 16 + col0;
      int rowB = mBase + wM + i * 16 + rsel;
#pragma unroll
      for (int r = 0; r < 8; ++r)
        C[(size_t)(rowB + r) * N + col] = (OutT)acc[i][j][r];
    }
}

// ---------------------------------------------------------------------------
// Flash attention with causal ALiBi bias.
// Grid: (SEQ/128, NHEADS, NBATCH), block 256 (8 waves).
// Wave w owns 16 q rows; 64-key K/V blocks staged in LDS (K via TDM when
// available, V transposed manually). Online softmax in f32.
// ---------------------------------------------------------------------------
__global__ __launch_bounds__(256) void mla_attn_wmma(
    const _Float16* __restrict__ Q, const _Float16* __restrict__ Km,
    const _Float16* __restrict__ Vm, _Float16* __restrict__ Om) {
  __shared__ alignas(16) _Float16 Ks[64 * 128];      // [key][dim]
  __shared__ alignas(16) _Float16 Vt[128 * 64];      // [dim][key]
  __shared__ alignas(16) _Float16 Ps[8 * 16 * 64];   // per-wave P tile

  const int tid  = threadIdx.x;
  const int lane = tid & 31;
  const int wave = tid >> 5;
  const int qBase = blockIdx.x * 128;
  const int head  = blockIdx.y;
  const int batch = blockIdx.z;
  const size_t tokBase = (size_t)batch * SEQ;
  const int hOff = head * HEAD_DIM;
  const int qRow0 = qBase + wave * 16;   // seq position of this wave's rows
  const int qRowMax = qRow0 + 15;
  const float scaleA = 0.08838834764831845f;  // 128^-0.5

  // Q fragments: 16 rows x 128 dims = 4 A-fragments (K chunks of 32).
  v16h qF[4];
#pragma unroll
  for (int c = 0; c < 4; ++c)
    qF[c] = load_fragA(Q + (tokBase + qRow0) * HIDDEN + hOff, HIDDEN, 0, c * 32, lane);

  float m_i[8], l_i[8];
  v8f zero = {};
  v8f o[8];
#pragma unroll
  for (int r = 0; r < 8; ++r) { m_i[r] = -1e30f; l_i[r] = 0.0f; }
#pragma unroll
  for (int d = 0; d < 8; ++d) o[d] = zero;

  _Float16* PsW = Ps + wave * 16 * 64;
  const int rsel = (lane >> 4) << 3;
  const int col0 = lane & 15;

  for (int kb = 0; kb < qBase + 128; kb += 64) {
#if HAVE_TDM
    // Stage K block with the Tensor Data Mover: 2-D tile, 128 x 64 f16 elems,
    // row stride HIDDEN. Descriptor per CDNA5 ISA D# groups 0/1.
    if (wave == 0) {
      const _Float16* gtile = Km + (tokBase + kb) * HIDDEN + hOff;
      unsigned long long ga = (unsigned long long)(size_t)gtile;
      unsigned ldsOff =
          (unsigned)(size_t)(__attribute__((address_space(3))) _Float16*)Ks;
      u32x4 g0;
      g0[0] = 1u;                                   // count=1, user descriptor
      g0[1] = ldsOff;                               // lds_addr (bytes)
      g0[2] = (unsigned)ga;                         // global_addr[31:0]
      g0[3] = (unsigned)((ga >> 32) & 0x1FFFFFFull) // global_addr[56:32]
              | (2u << 30);                         // type=2
      i32x8 g1;
      g1[0] = (int)(1u << 16);      // data_size = 2 bytes
      g1[1] = (int)(128u << 16);    // tensor_dim0 = 128 (tile-local, no OOB)
      g1[2] = (int)(64u << 16);     // tensor_dim1 = 64
      g1[3] = (int)(128u << 16);    // tile_dim0 = 128
      g1[4] = 64;                   // tile_dim1 = 64, tile_dim2 = 0
      g1[5] = HIDDEN;               // tensor_dim0_stride[31:0] = 2048
      g1[6] = 0;                    // stride[47:32] = 0, dim1_stride lo = 0
      g1[7] = 0;
      i32x4 gz = {0, 0, 0, 0};
      i32x8 gz8 = {0, 0, 0, 0, 0, 0, 0, 0};
      __builtin_amdgcn_tensor_load_to_lds(g0, g1, gz, gz, gz8, 0);
      __builtin_amdgcn_s_wait_tensorcnt(0);
    }
    // V block: transposed scatter (TDM cannot transpose).
#pragma unroll
    for (int t = 0; t < 4; ++t) {
      int chunk = tid + t * 256;     // 0..1023
      int row   = chunk >> 4;        // key 0..63
      int part  = (chunk & 15) * 8;  // dim offset
      union { uint4 u; _Float16 h[8]; } vv;
      vv.u = *(const uint4*)(Vm + (tokBase + kb + row) * HIDDEN + hOff + part);
#pragma unroll
      for (int d2 = 0; d2 < 8; ++d2) Vt[(part + d2) * 64 + row] = vv.h[d2];
    }
#else
    // Fallback: stage K (row-major) and V (transposed) manually.
#pragma unroll
    for (int t = 0; t < 4; ++t) {
      int chunk = tid + t * 256;
      int row   = chunk >> 4;
      int part  = (chunk & 15) * 8;
      *(uint4*)(Ks + row * 128 + part) =
          *(const uint4*)(Km + (tokBase + kb + row) * HIDDEN + hOff + part);
      union { uint4 u; _Float16 h[8]; } vv;
      vv.u = *(const uint4*)(Vm + (tokBase + kb + row) * HIDDEN + hOff + part);
#pragma unroll
      for (int d2 = 0; d2 < 8; ++d2) Vt[(part + d2) * 64 + row] = vv.h[d2];
    }
#endif
    __syncthreads();

    if (kb <= qRowMax) {  // skip fully-masked blocks (no barrier inside)
      // S = Q @ K^T : 16 x 64 scores = 4 N-subtiles x 4 K-dim chunks.
      v8f s[4];
      for (int n = 0; n < 4; ++n) s[n] = zero;
#pragma unroll
      for (int c = 0; c < 4; ++c) {
#pragma unroll
        for (int n = 0; n < 4; ++n) {
          v16h bF = load_fragB(Ks, 128, n * 16, c * 32, lane);
          s[n] = __builtin_amdgcn_wmma_f32_16x16x32_f16(
              false, qF[c], false, bF, (short)0, s[n], false, false);
        }
      }
      // Scale + causal ALiBi bias.
      float pb[4][8];
#pragma unroll
      for (int n = 0; n < 4; ++n) {
        int j = kb + n * 16 + col0;
#pragma unroll
        for (int r = 0; r < 8; ++r) {
          int i = qRow0 + r + rsel;
          float val = s[n][r] * scaleA;
          pb[n][r] = (j <= i) ? (val - SLOPE * (float)(i - j)) : -1e30f;
        }
      }
      // Online softmax per row (16 lanes of each half-wave hold one row).
#pragma unroll
      for (int r = 0; r < 8; ++r) {
        float mx = fmaxf(fmaxf(pb[0][r], pb[1][r]), fmaxf(pb[2][r], pb[3][r]));
        mx = fmaxf(mx, __shfl_xor(mx, 1, 32));
        mx = fmaxf(mx, __shfl_xor(mx, 2, 32));
        mx = fmaxf(mx, __shfl_xor(mx, 4, 32));
        mx = fmaxf(mx, __shfl_xor(mx, 8, 32));
        float mn = fmaxf(m_i[r], mx);
        float sc = __expf(m_i[r] - mn);
        m_i[r] = mn;
        float rs = 0.0f;
#pragma unroll
        for (int n = 0; n < 4; ++n) { pb[n][r] = __expf(pb[n][r] - mn); rs += pb[n][r]; }
        rs += __shfl_xor(rs, 1, 32);
        rs += __shfl_xor(rs, 2, 32);
        rs += __shfl_xor(rs, 4, 32);
        rs += __shfl_xor(rs, 8, 32);
        l_i[r] = l_i[r] * sc + rs;
#pragma unroll
        for (int d = 0; d < 8; ++d) o[d][r] *= sc;
      }
      // Re-layout P (accumulator layout) -> A-fragment layout through LDS.
#pragma unroll
      for (int n = 0; n < 4; ++n)
#pragma unroll
        for (int r = 0; r < 8; ++r)
          PsW[(r + rsel) * 64 + n * 16 + col0] = (_Float16)pb[n][r];
      // O += P @ V : K = 64 keys (2 chunks), N = 128 dims (8 subtiles).
#pragma unroll
      for (int c2 = 0; c2 < 2; ++c2) {
        v16h aP = load_fragA(PsW, 64, 0, c2 * 32, lane);
#pragma unroll
        for (int d = 0; d < 8; ++d) {
          v16h bV = load_fragB(Vt, 64, d * 16, c2 * 32, lane);
          o[d] = __builtin_amdgcn_wmma_f32_16x16x32_f16(
              false, aP, false, bV, (short)0, o[d], false, false);
        }
      }
    }
    __syncthreads();
  }

  // Normalize and store (f16, feeds the output projection GEMM).
#pragma unroll
  for (int d = 0; d < 8; ++d) {
    int col = hOff + d * 16 + col0;
#pragma unroll
    for (int r = 0; r < 8; ++r) {
      float inv = 1.0f / l_i[r];
      Om[(tokBase + qRow0 + r + rsel) * HIDDEN + col] = (_Float16)(o[d][r] * inv);
    }
  }
}

// ---------------------------------------------------------------------------
__global__ void cvt_f32_to_f16(const float* __restrict__ in,
                               _Float16* __restrict__ out, size_t n) {
  size_t i = (size_t)blockIdx.x * blockDim.x + threadIdx.x;
  if (i < n) out[i] = (_Float16)in[i];
}

// ---------------------------------------------------------------------------
extern "C" void kernel_launch(void* const* d_in, const int* in_sizes, int n_in,
                              void* d_out, int out_size, void* d_ws, size_t ws_size,
                              hipStream_t stream) {
  const float* hs   = (const float*)d_in[0];
  const float* wdkv = (const float*)d_in[1];
  const float* wq   = (const float*)d_in[2];
  const float* wuk  = (const float*)d_in[3];
  const float* wuv  = (const float*)d_in[4];
  const float* wo   = (const float*)d_in[5];
  float* out = (float*)d_out;

  const size_t T = (size_t)NBATCH * SEQ;  // 4096 tokens

  char* ws = (char*)d_ws;
  size_t off = 0;
  auto alloc = [&](size_t halves) {
    _Float16* p = (_Float16*)(ws + off);
    off += ((halves * sizeof(_Float16) + 255) / 256) * 256;
    return p;
  };
  _Float16* h16    = alloc(T * HIDDEN);
  _Float16* wq16   = alloc((size_t)HIDDEN * HIDDEN);
  _Float16* wdkv16 = alloc((size_t)KV_DIM * HIDDEN);
  _Float16* wuk16  = alloc((size_t)HIDDEN * KV_DIM);
  _Float16* wuv16  = alloc((size_t)HIDDEN * KV_DIM);
  _Float16* wo16   = alloc((size_t)HIDDEN * HIDDEN);
  _Float16* ckv16  = alloc(T * KV_DIM);
  _Float16* q16    = alloc(T * HIDDEN);
  _Float16* k16    = alloc(T * HIDDEN);
  _Float16* v16    = alloc(T * HIDDEN);
  _Float16* a16    = alloc(T * HIDDEN);

  auto cvt = [&](const float* src, _Float16* dst, size_t n) {
    cvt_f32_to_f16<<<dim3((unsigned)((n + 255) / 256)), dim3(256), 0, stream>>>(src, dst, n);
  };
  cvt(hs,   h16,    T * HIDDEN);
  cvt(wdkv, wdkv16, (size_t)KV_DIM * HIDDEN);
  cvt(wq,   wq16,   (size_t)HIDDEN * HIDDEN);
  cvt(wuk,  wuk16,  (size_t)HIDDEN * KV_DIM);
  cvt(wuv,  wuv16,  (size_t)HIDDEN * KV_DIM);
  cvt(wo,   wo16,   (size_t)HIDDEN * HIDDEN);

  // q = h @ w_q^T            [4096,2048] x [2048,2048]^T
  gemm_f16_wmma<_Float16><<<dim3(HIDDEN / 128, T / 128), 256, 0, stream>>>(
      h16, wq16, q16, (int)T, HIDDEN, HIDDEN);
  // c_kv = h @ w_dkv^T       [4096,2048] x [128,2048]^T
  gemm_f16_wmma<_Float16><<<dim3(KV_DIM / 128, T / 128), 256, 0, stream>>>(
      h16, wdkv16, ckv16, (int)T, KV_DIM, HIDDEN);
  // k = c_kv @ w_uk^T        [4096,128] x [2048,128]^T
  gemm_f16_wmma<_Float16><<<dim3(HIDDEN / 128, T / 128), 256, 0, stream>>>(
      ckv16, wuk16, k16, (int)T, HIDDEN, KV_DIM);
  // v = c_kv @ w_uv^T
  gemm_f16_wmma<_Float16><<<dim3(HIDDEN / 128, T / 128), 256, 0, stream>>>(
      ckv16, wuv16, v16, (int)T, HIDDEN, KV_DIM);
  // flash attention with causal ALiBi
  mla_attn_wmma<<<dim3(SEQ / 128, NHEADS, NBATCH), 256, 0, stream>>>(q16, k16, v16, a16);
  // out = attn @ w_o^T  (f32 output)
  gemm_f16_wmma<float><<<dim3(HIDDEN / 128, T / 128), 256, 0, stream>>>(
      a16, wo16, out, (int)T, HIDDEN, HIDDEN);
}
